// ProductMonoidHead_84293028151341
// MI455X (gfx1250) — 16-bit and 8-bit matrix operand order repaired and checked by reading, compiled, not run on hardware
//
#include <hip/hip_runtime.h>

typedef __attribute__((ext_vector_type(16))) _Float16 v16h;
typedef __attribute__((ext_vector_type(8)))  float    v8f;
typedef __attribute__((ext_vector_type(8)))  int      v8i;
typedef __attribute__((ext_vector_type(2)))  _Float16 h2;

namespace {
constexpr int kB = 8, kN = 1024, kM = 1024, kD = 1024, kH = 128;
constexpr int kRows = kB * kN;
constexpr float kScale = 0.17677669529663687f;
constexpr int kScStride = 1028;
constexpr size_t kAttnLds = (size_t)16 * kScStride * sizeof(float) + 16 * 16 * sizeof(float);

union FragH { v16h v; _Float16 h[16]; unsigned u[8]; };
union FragF { v8f v;  float f[8]; };
union FragI { v8i v;  unsigned u[8]; int i[8]; };
union UH2   { unsigned u; h2 h; };
}

__device__ __forceinline__ int koffA(int v, int hi) {
  return (v < 4) ? (hi * 8 + 2 * v) : (16 + hi * 8 + 2 * (v - 4));
}

__global__ void wcvt_kernel(const float* __restrict__ src,
                            _Float16* __restrict__ dst, int J) {
  int idx = blockIdx.x * blockDim.x + threadIdx.x;
  if (idx >= J * kD) return;
  int j = idx / kD, d = idx - j * kD;
  dst[j * kD + d] = (_Float16)src[d * J + j];
}

__global__ void __launch_bounds__(128) proj_qk_kernel(
    const float* __restrict__ X,
    const _Float16* __restrict__ WTb, const _Float16* __restrict__ WTt,
    const _Float16* __restrict__ WTr,
    signed char* __restrict__ outS, _Float16* __restrict__ outT,
    _Float16* __restrict__ outR) {
  const int lane = threadIdx.x & 31;
  const int wid  = threadIdx.x >> 5;
  const int hi   = lane >> 4;
  const int ln   = lane & 15;
  const int rowbase = (blockIdx.x * 4 + wid) * 16;

  const unsigned* Ws[3] = {(const unsigned*)WTb, (const unsigned*)WTt,
                           (const unsigned*)WTr};
  v8f acc[3][2] = {};

  const float* xrow = X + (size_t)(rowbase + ln) * kD;

  for (int kc = 0; kc < kD / 32; ++kc) {
    const int k0 = kc * 32;
    FragH a;
#pragma unroll
    for (int v = 0; v < 8; ++v) {
      int k = k0 + 16 * hi + 2 * v;
      a.h[2 * v]     = (_Float16)xrow[k];
      a.h[2 * v + 1] = (_Float16)xrow[k + 1];
    }
    const int kb = kc * 16 + hi * 8;
#pragma unroll
    for (int w = 0; w < 3; ++w) {
#pragma unroll
      for (int s = 0; s < 2; ++s) {
        FragH bf;
#pragma unroll
        for (int v = 0; v < 8; ++v) bf.u[v] = Ws[w][(s * 16 + ln) * 512 + kb + v];
        acc[w][s] = __builtin_amdgcn_wmma_f32_16x16x32_f16(
            false, a.v, false, bf.v, (short)0, acc[w][s], false, false);
      }
    }
  }

#pragma unroll
  for (int s = 0; s < 2; ++s) {
    FragF fb, ft, fr;
    fb.v = acc[0][s]; ft.v = acc[1][s]; fr.v = acc[2][s];
#pragma unroll
    for (int v = 0; v < 8; ++v) {
      int r = rowbase + v + 8 * hi;
      int c = s * 16 + ln;
      outS[r * 32 + c] = (signed char)(fb.f[v] > 0.0f ? 1 : -1);
      outT[r * 32 + c] = (_Float16)ft.f[v];
      outR[r * 32 + c] = (_Float16)fr.f[v];
    }
  }
}

__global__ void __launch_bounds__(128) proj_v_kernel(
    const float* __restrict__ V, const _Float16* __restrict__ WTv,
    _Float16* __restrict__ VpT) {
  const int lane = threadIdx.x & 31;
  const int wid  = threadIdx.x >> 5;
  const int hi   = lane >> 4;
  const int ln   = lane & 15;
  const int rowbase = (blockIdx.x * 4 + wid) * 16;

  const unsigned* Wu = (const unsigned*)WTv;
  v8f acc[8] = {};
  const float* xrow = V + (size_t)(rowbase + ln) * kD;

  for (int kc = 0; kc < kD / 32; ++kc) {
    const int k0 = kc * 32;
    FragH a;
#pragma unroll
    for (int v = 0; v < 8; ++v) {
      int k = k0 + 16 * hi + 2 * v;
      a.h[2 * v]     = (_Float16)xrow[k];
      a.h[2 * v + 1] = (_Float16)xrow[k + 1];
    }
    const int kb = kc * 16 + hi * 8;
#pragma unroll
    for (int t = 0; t < 8; ++t) {
      FragH bf;
#pragma unroll
      for (int v = 0; v < 8; ++v) bf.u[v] = Wu[(t * 16 + ln) * 512 + kb + v];
      acc[t] = __builtin_amdgcn_wmma_f32_16x16x32_f16(
          false, a.v, false, bf.v, (short)0, acc[t], false, false);
    }
  }
#pragma unroll
  for (int t = 0; t < 8; ++t) {
    FragF f; f.v = acc[t];
#pragma unroll
    for (int v = 0; v < 8; ++v) {
      int r = rowbase + v + 8 * hi;
      int bb = r >> 10, m = r & 1023;
      VpT[((size_t)(bb * kH + t * 16 + ln)) * kM + m] = (_Float16)f.f[v];
    }
  }
}

__global__ void __launch_bounds__(256) attn_kernel(
    const _Float16* __restrict__ Qr, const _Float16* __restrict__ Qt,
    const signed char* __restrict__ Qs, const _Float16* __restrict__ Kr,
    const _Float16* __restrict__ Kt, const signed char* __restrict__ Ks,
    const _Float16* __restrict__ VpT, const float* __restrict__ fusion,
    const int* __restrict__ mask, float* __restrict__ out,
    float* __restrict__ attn) {
  extern __shared__ char smem[];
  float* sc = (float*)smem;
  float* red = (float*)(smem + (size_t)16 * kScStride * 4);
  unsigned* qtl = (unsigned*)red;

  const int tid  = threadIdx.x;
  const int lane = tid & 31;
  const int wid  = tid >> 5;
  const int hi   = lane >> 4;
  const int ln   = lane & 15;
  const int b     = blockIdx.x >> 6;
  const int nbase = (blockIdx.x & 63) * 16;

  float f0 = fusion[0], f1 = fusion[1], f2 = fusion[2];
  float fm = fmaxf(f0, fmaxf(f1, f2));
  float e0 = __expf(f0 - fm), e1 = __expf(f1 - fm), e2 = __expf(f2 - fm);
  float einv = 1.0f / (e0 + e1 + e2);
  float w0 = e0 * einv, w1 = e1 * einv, w2 = e2 * einv;

  {
    int r = tid >> 4, p = tid & 15;
    const unsigned* Qtu = (const unsigned*)Qt;
    qtl[r * 16 + p] = Qtu[(b * kN + nbase + r) * 16 + p];
  }

  FragH aR;
  {
    const unsigned* Qru = (const unsigned*)Qr;
    int arow = b * kN + nbase + ln;
#pragma unroll
    for (int v = 0; v < 8; ++v) aR.u[v] = Qru[arow * 16 + hi * 8 + v];
  }
  FragI aB;
  {
    const unsigned* Qsu = (const unsigned*)Qs;
    int arow = b * kN + nbase + ln;
    int o = hi ? 2 : 0;
    aB.u[0] = Qsu[arow * 8 + 0 + o];
    aB.u[1] = Qsu[arow * 8 + 1 + o];
    aB.u[2] = Qsu[arow * 8 + 4 + o];
    aB.u[3] = Qsu[arow * 8 + 5 + o];
    aB.u[4] = aB.u[5] = aB.u[6] = aB.u[7] = 0;
  }
  __syncthreads();

  const unsigned* Kru = (const unsigned*)Kr;
  const unsigned* Ktu = (const unsigned*)Kt;
  const unsigned* Ksu = (const unsigned*)Ks;

  for (int mt = wid; mt < 64; mt += 8) {
    const int m0 = mt * 16;
    const int bcol = b * kM + m0 + ln;

    FragH bR;
#pragma unroll
    for (int v = 0; v < 8; ++v) bR.u[v] = Kru[bcol * 16 + hi * 8 + v];

    FragI bB;
#pragma unroll
    for (int i = 0; i < 4; ++i) bB.u[i] = Ksu[bcol * 8 + (hi ? 2 : 0) + (i & 1) + (i >> 1) * 4];
    bB.u[4] = bB.u[5] = bB.u[6] = bB.u[7] = 0;

    UH2 kt[16];
#pragma unroll
    for (int j = 0; j < 16; ++j) kt[j].u = Ktu[bcol * 16 + j];

    v8f cz = {};
    FragF rr;
    rr.v = __builtin_amdgcn_wmma_f32_16x16x32_f16(false, aR.v, false, bR.v,
                                                  (short)0, cz, false, false);
    v8i ciz = {};
    FragI ri;
    ri.v = __builtin_amdgcn_wmma_i32_16x16x64_iu8(true, aB.v, true, bB.v, ciz,
                                                  false, false);

#pragma unroll
    for (int v = 0; v < 8; ++v) {
      const int M = v + 8 * hi;
      h2 acc2; acc2.x = (_Float16)(-60000.0f); acc2.y = acc2.x;
#pragma unroll
      for (int j = 0; j < 16; ++j) {
        UH2 q; q.u = qtl[M * 16 + j];
        h2 s2 = q.h + kt[j].h;
        acc2 = __builtin_elementwise_max(acc2, s2);
      }
      float tv = fmaxf((float)acc2.x, (float)acc2.y);
      float matches = 0.5f * (float)(ri.i[v] + 32);
      float s = kScale * (w0 * matches + w1 * tv + w2 * rr.f[v]);
      int mk = mask[(b * kN + nbase + M) * kM + m0 + ln];
      if (mk == 0) s = -__builtin_inff();
      sc[M * kScStride + m0 + ln] = s;
    }
  }
  __syncthreads();

  {
    int r = tid >> 4, l = tid & 15;
    float* row = sc + r * kScStride;
    float pm = -__builtin_inff();
    for (int i = 0; i < 64; ++i) pm = fmaxf(pm, row[l + 16 * i]);
    red[r * 16 + l] = pm;
    __syncthreads();
    float rm = -__builtin_inff();
#pragma unroll
    for (int i = 0; i < 16; ++i) rm = fmaxf(rm, red[r * 16 + i]);
    __syncthreads();
    float ps = 0.0f;
    for (int i = 0; i < 64; ++i) {
      int m = l + 16 * i;
      float e = __expf(row[m] - rm);
      row[m] = e;
      ps += e;
    }
    red[r * 16 + l] = ps;
    __syncthreads();
    float rs = 0.0f;
#pragma unroll
    for (int i = 0; i < 16; ++i) rs += red[r * 16 + i];
    float inv = 1.0f / rs;
    float* att = attn + (size_t)(b * kN + nbase + r) * kM;
    for (int i = 0; i < 64; ++i) {
      int m = l + 16 * i;
      float a = row[m] * inv;
      row[m] = a;
      att[m] = a;
    }
  }
  __syncthreads();

  {
    const int j0 = wid * 16;
    const unsigned* Vu = (const unsigned*)VpT;
    const int vrow = b * kH + j0 + ln;
    v8f acc = {};
    for (int kc = 0; kc < kM / 32; ++kc) {
      const int k0 = kc * 32;
      FragH a;
#pragma unroll
      for (int v = 0; v < 8; ++v) {
        int k = k0 + 16 * hi + 2 * v;
        a.h[2 * v]     = (_Float16)sc[ln * kScStride + k];
        a.h[2 * v + 1] = (_Float16)sc[ln * kScStride + k + 1];
      }
      FragH bf;
#pragma unroll
      for (int v = 0; v < 8; ++v) bf.u[v] = Vu[vrow * 512 + kc * 16 + hi * 8 + v];
      acc = __builtin_amdgcn_wmma_f32_16x16x32_f16(false, a.v, false, bf.v,
                                                   (short)0, acc, false, false);
    }
    FragF o; o.v = acc;
#pragma unroll
    for (int v = 0; v < 8; ++v) {
      int M = v + 8 * hi;
      out[(size_t)(b * kN + nbase + M) * kH + j0 + ln] = o.f[v];
    }
  }
}

extern "C" void kernel_launch(void* const* d_in, const int* in_sizes, int n_in,
                              void* d_out, int out_size, void* d_ws,
                              size_t ws_size, hipStream_t stream) {
  const float* Q    = (const float*)d_in[0];
  const float* K    = (const float*)d_in[1];
  const float* V    = (const float*)d_in[2];
  const float* Wqb  = (const float*)d_in[3];
  const float* Wkb  = (const float*)d_in[4];
  const float* Wqt  = (const float*)d_in[5];
  const float* Wkt  = (const float*)d_in[6];
  const float* Wqr  = (const float*)d_in[7];
  const float* Wkr  = (const float*)d_in[8];
  const float* Wv   = (const float*)d_in[9];
  const float* fus  = (const float*)d_in[10];
  const int*   mask = (const int*)d_in[11];

  float* out  = (float*)d_out;
  float* attn = out + (size_t)kB * kN * kH;

  char* ws = (char*)d_ws;
  size_t o = 0;
  auto take = [&](size_t bytes) { char* p = ws + o; o += bytes; return p; };
  _Float16* WTqb = (_Float16*)take(32 * kD * 2);
  _Float16* WTqt = (_Float16*)take(32 * kD * 2);
  _Float16* WTqr = (_Float16*)take(32 * kD * 2);
  _Float16* WTkb = (_Float16*)take(32 * kD * 2);
  _Float16* WTkt = (_Float16*)take(32 * kD * 2);
  _Float16* WTkr = (_Float16*)take(32 * kD * 2);
  _Float16* WTv  = (_Float16*)take((size_t)kH * kD * 2);
  _Float16* Qrc  = (_Float16*)take((size_t)kRows * 32 * 2);
  _Float16* Qtc  = (_Float16*)take((size_t)kRows * 32 * 2);
  signed char* Qsc = (signed char*)take((size_t)kRows * 32);
  _Float16* Krc  = (_Float16*)take((size_t)kRows * 32 * 2);
  _Float16* Ktc  = (_Float16*)take((size_t)kRows * 32 * 2);
  signed char* Ksc = (signed char*)take((size_t)kRows * 32);
  _Float16* VpT  = (_Float16*)take((size_t)kB * kH * kM * 2);
  (void)ws_size; (void)in_sizes; (void)n_in; (void)out_size;

  wcvt_kernel<<<(32 * kD + 255) / 256, 256, 0, stream>>>(Wqb, WTqb, 32);
  wcvt_kernel<<<(32 * kD + 255) / 256, 256, 0, stream>>>(Wqt, WTqt, 32);
  wcvt_kernel<<<(32 * kD + 255) / 256, 256, 0, stream>>>(Wqr, WTqr, 32);
  wcvt_kernel<<<(32 * kD + 255) / 256, 256, 0, stream>>>(Wkb, WTkb, 32);
  wcvt_kernel<<<(32 * kD + 255) / 256, 256, 0, stream>>>(Wkt, WTkt, 32);
  wcvt_kernel<<<(32 * kD + 255) / 256, 256, 0, stream>>>(Wkr, WTkr, 32);
  wcvt_kernel<<<(kH * kD + 255) / 256, 256, 0, stream>>>(Wv, WTv, kH);

  proj_qk_kernel<<<kRows / 64, 128, 0, stream>>>(Q, WTqb, WTqt, WTqr, Qsc, Qtc, Qrc);
  proj_qk_kernel<<<kRows / 64, 128, 0, stream>>>(K, WTkb, WTkt, WTkr, Ksc, Ktc, Krc);
  proj_v_kernel<<<kRows / 64, 128, 0, stream>>>(V, WTv, VpT);

  attn_kernel<<<kB * (kN / 16), 256, kAttnLds, stream>>>(
      Qrc, Qtc, Qsc, Krc, Ktc, Ksc, VpT, fus, mask, out, attn);
}
